// VisualEncoderDecouplingFusion_51049981280483
// MI455X (gfx1250) — matrix operand order repaired and checked by reading, not compiled, not run on hardware
//
#include <hip/hip_runtime.h>
#include <hip/hip_bf16.h>
#include <math.h>

// ---------------------------------------------------------------------------
// Types for CDNA5 WMMA / TDM
// ---------------------------------------------------------------------------
typedef __attribute__((ext_vector_type(16))) __bf16 v16bf;
typedef __attribute__((ext_vector_type(8)))  float  v8f;
typedef __attribute__((ext_vector_type(4)))  unsigned int u32x4;
typedef __attribute__((ext_vector_type(8)))  unsigned int u32x8;

// ---------------------------------------------------------------------------
// Helpers
// ---------------------------------------------------------------------------
__device__ __forceinline__ unsigned short f2bf(float f) {
    unsigned int u = __float_as_uint(f);
    unsigned int r = (u + 0x7FFFu + ((u >> 16) & 1u)) >> 16;
    return (unsigned short)r;
}

__device__ __forceinline__ unsigned int fkey(float f) {
    unsigned int u = __float_as_uint(f);
    return (u & 0x80000000u) ? ~u : (u | 0x80000000u);
}

__device__ __forceinline__ float gelu_exact(float x) {
    return 0.5f * x * (1.0f + erff(x * 0.70710678118654752f));
}

// ---------------------------------------------------------------------------
// Naive direct convolution (small share of total FLOPs)
// ---------------------------------------------------------------------------
__global__ void conv2d_kernel(const float* __restrict__ in, const float* __restrict__ w,
                              const float* __restrict__ bias, float* __restrict__ out,
                              int Bn, int Cin, int Hin, int Win,
                              int Cout, int Hout, int Wout,
                              int kh, int kw, int stride, int pad, int relu) {
    int idx = blockIdx.x * blockDim.x + threadIdx.x;
    int total = Bn * Cout * Hout * Wout;
    if (idx >= total) return;
    int wo = idx % Wout; int t = idx / Wout;
    int ho = t % Hout;   t /= Hout;
    int co = t % Cout;   int b = t / Cout;
    float acc = bias[co];
    int hbase = ho * stride - pad;
    int wbase = wo * stride - pad;
    for (int ci = 0; ci < Cin; ++ci) {
        const float* inp = in + ((b * Cin + ci) * Hin) * Win;
        const float* wp  = w + ((co * Cin + ci) * kh) * kw;
        for (int r = 0; r < kh; ++r) {
            int hi = hbase + r;
            if ((unsigned)hi >= (unsigned)Hin) continue;
            for (int c = 0; c < kw; ++c) {
                int wi = wbase + c;
                if ((unsigned)wi >= (unsigned)Win) continue;
                acc += inp[hi * Win + wi] * wp[r * kw + c];
            }
        }
    }
    if (relu) acc = fmaxf(acc, 0.0f);
    out[idx] = acc;
}

// ---------------------------------------------------------------------------
// Elementwise cast f32 -> bf16 (stored as raw ushort)
// ---------------------------------------------------------------------------
__global__ void cast_bf16_kernel(const float* __restrict__ src, unsigned short* __restrict__ dst, int n) {
    int idx = blockIdx.x * blockDim.x + threadIdx.x;
    if (idx < n) dst[idx] = f2bf(src[idx]);
}

// NCHW [B,D,32,32] -> [B*1024, D] (tokens-major) with bf16 cast
__global__ void emb_transpose_cast_kernel(const float* __restrict__ x, unsigned short* __restrict__ dst,
                                          int Bn, int D, int HW) {
    int idx = blockIdx.x * blockDim.x + threadIdx.x;
    int total = Bn * HW * D;
    if (idx >= total) return;
    int d = idx % D;
    int m = idx / D;            // b*HW + n
    int b = m / HW;
    int n = m % HW;
    float v = x[((b * D) + d) * HW + n];
    dst[idx] = f2bf(v);
}

__global__ void cbsq_kernel(const float* __restrict__ cb, float* __restrict__ cbsq, int K, int D) {
    int k = blockIdx.x * blockDim.x + threadIdx.x;
    if (k >= K) return;
    const float* row = cb + (size_t)k * D;
    float s = 0.f;
    for (int d = 0; d < D; ++d) s += row[d] * row[d];
    cbsq[k] = s;
}

__global__ void amin_init_kernel(unsigned long long* __restrict__ a, int n) {
    int idx = blockIdx.x * blockDim.x + threadIdx.x;
    if (idx < n) a[idx] = 0xFFFFFFFFFFFFFFFFull;
}

__global__ void gather_cast_kernel(const unsigned long long* __restrict__ amin,
                                   const float* __restrict__ cb,
                                   unsigned short* __restrict__ quant, int M, int D) {
    int idx = blockIdx.x * blockDim.x + threadIdx.x;
    if (idx >= M * D) return;
    int m = idx / D;
    int d = idx % D;
    unsigned int k = (unsigned int)(amin[m] & 0xFFFFFFFFull);
    quant[idx] = f2bf(cb[(size_t)k * D + d]);
}

// ---------------------------------------------------------------------------
// WMMA GEMM: C[M,N] = A[M,K] * B[N,K]^T   (A,B bf16 row-major, acc f32)
// Block tile 128x128, BK=32, 256 threads = 8 waves, wave tile 64x32.
// A tile is staged into LDS by the Tensor Data Mover (TDM): wave 0 builds a
// 2-group D# descriptor in SGPRs and issues tensor_load_to_lds; LDS row
// padding (32 data shorts + 8 pad shorts) is encoded via pad_interval=16 DW,
// pad_amount=4 DW. B tile is transposed manually into LDS (TDM cannot
// transpose elements).
// MODE 0: out_f32[m*N+n] = acc + bias[n]
// MODE 1: out_bf16[m*N+n] = bf16(gelu(acc + bias[n]))
// MODE 2: VQ argmin epilogue: score = cbsq[n] - 2*acc, packed atomicMin per row
// ---------------------------------------------------------------------------
template<int MODE>
__global__ __launch_bounds__(256)
void gemm_wmma_kernel(const unsigned short* __restrict__ A,
                      const unsigned short* __restrict__ B,
                      const float* __restrict__ bias,
                      float* __restrict__ outF,
                      unsigned short* __restrict__ outB,
                      const float* __restrict__ cbsq,
                      unsigned long long* __restrict__ amin,
                      int M, int N, int K) {
    __shared__ unsigned short As[128][40];   // [m][k], 32 data + 8 pad shorts/row
    __shared__ unsigned short Bs[32][136];   // [k][n], padded rows (272 B)

    const int tid   = threadIdx.x;
    const int lane  = tid & 31;
    const int wave  = tid >> 5;
    const int wm    = wave >> 2;          // 0..1  (64 rows each)
    const int wn    = wave & 3;           // 0..3  (32 cols each)
    const int l16   = lane & 15;
    const int lhalf = lane >> 4;

    const int mBase = blockIdx.y * 128;
    const int nBase = blockIdx.x * 128;

    const int r    = tid >> 1;            // 0..127 row within tile
    const int part = tid & 1;             // 0/1 half of a 32-wide K slice

    // Uniform pieces of the TDM descriptor (valid for all K-steps).
    const unsigned ldsA = (unsigned)(size_t)(&As[0][0]);
    const unsigned long long gbaseA =
        (unsigned long long)(size_t)A + ((unsigned long long)mBase * (unsigned)K) * 2ull;

    v8f acc[4][2];
#pragma unroll
    for (int i = 0; i < 4; ++i)
#pragma unroll
        for (int j = 0; j < 2; ++j)
            acc[i][j] = (v8f){0.f, 0.f, 0.f, 0.f, 0.f, 0.f, 0.f, 0.f};

    for (int k0 = 0; k0 < K; k0 += 32) {
        // ---- B global load (16 bf16 = 32 B per thread) ----
        const unsigned short* gb = B + (size_t)(nBase + r) * K + k0 + part * 16;
        union { uint4 q[2]; unsigned short s[16]; } tb;
        tb.q[0] = ((const uint4*)gb)[0];
        tb.q[1] = ((const uint4*)gb)[1];
        if (k0 + 32 < K) __builtin_prefetch(gb + 32, 0, 1);

        __syncthreads();   // previous compute done before LDS is overwritten

        // ---- A tile via Tensor Data Mover (one descriptor per workgroup) ----
        if (wave == 0) {
            unsigned long long ga = gbaseA + (unsigned long long)k0 * 2ull;
            u32x4 g0 = {
                1u,                                    // count=1, user mode
                ldsA,                                  // lds_addr
                (unsigned)ga,                          // global_addr[31:0]
                ((unsigned)(ga >> 32) & 0x01FFFFFFu) | 0x80000000u  // addr[56:32], type=2
            };
            u32x8 g1 = {
                (1u << 16) | (1u << 20) | (3u << 22) | (3u << 25), // data_size=2B, pad_en, interval=16DW, amount=4DW
                ((unsigned)K & 0xFFFFu) << 16,                     // abar=0 | tensor_dim0[15:0]
                (((unsigned)K >> 16) & 0xFFFFu) | (((unsigned)M & 0xFFFFu) << 16), // dim0 hi | dim1 lo
                (((unsigned)M >> 16) & 0xFFFFu) | (32u << 16),     // dim1 hi | tile_dim0=32
                128u,                                              // tile_dim1=128, tile_dim2=0
                (unsigned)K,                                       // dim0_stride[31:0]
                0u,                                                // dim0_stride hi | dim1_stride lo
                0u                                                 // dim1_stride hi
            };
            asm volatile("tensor_load_to_lds %0, %1" :: "s"(g0), "s"(g1) : "memory");
        }

        // ---- B: transpose [n][k] -> Bs[k][n] ----
#pragma unroll
        for (int e = 0; e < 16; ++e)
            Bs[part * 16 + e][r] = tb.s[e];

        if (wave == 0) __builtin_amdgcn_s_wait_tensorcnt(0);
        __syncthreads();

        // ---- B fragments: lane%16 = N index, element e -> K = (e&7) + 8*lhalf + 16*(e>>3) ----
        v16bf bfrag[2];
#pragma unroll
        for (int js = 0; js < 2; ++js) {
            union { v16bf v; unsigned short s[16]; } t;
            const unsigned short* brow = &Bs[0][wn * 32 + js * 16 + l16];
#pragma unroll
            for (int j = 0; j < 16; ++j)
                t.s[j] = brow[((j & 7) + (lhalf << 3) + ((j >> 3) << 4)) * 136];
            bfrag[js] = t.v;
        }

        // ---- A fragments + WMMA ----
#pragma unroll
        for (int i = 0; i < 4; ++i) {
            union { v16bf v; unsigned int u[8]; } ta;
            const unsigned short* arow = &As[wm * 64 + i * 16 + l16][0];
#pragma unroll
            for (int j = 0; j < 8; ++j) {
                int kk = ((j & 3) << 1) + ((j >> 2) << 4) + (lhalf << 3);
                ta.u[j] = *(const unsigned int*)(arow + kk);
            }
            acc[i][0] = __builtin_amdgcn_wmma_f32_16x16x32_bf16(
                false, ta.v, false, bfrag[0], (short)0, acc[i][0], false, false);
            acc[i][1] = __builtin_amdgcn_wmma_f32_16x16x32_bf16(
                false, ta.v, false, bfrag[1], (short)0, acc[i][1], false, false);
        }
    }

    // ---- epilogue ----
    if (MODE == 0 || MODE == 1) {
#pragma unroll
        for (int i = 0; i < 4; ++i) {
#pragma unroll
            for (int j = 0; j < 2; ++j) {
                int n = nBase + wn * 32 + j * 16 + l16;
                float bv = bias[n];
#pragma unroll
                for (int rr = 0; rr < 8; ++rr) {
                    int m = mBase + wm * 64 + i * 16 + rr + 8 * lhalf;
                    float v = acc[i][j][rr] + bv;
                    if (MODE == 0) {
                        outF[(size_t)m * N + n] = v;
                    } else {
                        outB[(size_t)m * N + n] = f2bf(gelu_exact(v));
                    }
                }
            }
        }
    } else {  // MODE 2: fused VQ distance argmin
        int n0g = nBase + wn * 32 + l16;
        int n1g = n0g + 16;
        float c0 = cbsq[n0g];
        float c1 = cbsq[n1g];
#pragma unroll
        for (int i = 0; i < 4; ++i) {
#pragma unroll
            for (int rr = 0; rr < 8; ++rr) {
                float s0 = c0 - 2.f * acc[i][0][rr];
                float s1 = c1 - 2.f * acc[i][1][rr];
                float s = s0; int n = n0g;
                if (s1 < s || (s1 == s && n1g < n)) { s = s1; n = n1g; }
#pragma unroll
                for (int off = 1; off < 16; off <<= 1) {
                    float s2 = __shfl_xor(s, off, 32);
                    int   n2 = __shfl_xor(n, off, 32);
                    if (s2 < s || (s2 == s && n2 < n)) { s = s2; n = n2; }
                }
                if (l16 == 0) {
                    int m = mBase + wm * 64 + i * 16 + rr + 8 * lhalf;
                    unsigned long long key =
                        ((unsigned long long)fkey(s) << 32) | (unsigned int)n;
                    atomicMin(&amin[m], key);
                }
            }
        }
    }
}

// ---------------------------------------------------------------------------
// LayerNorm over rows of length L, scattered into the concat output layout.
// blockIdx.x = token m in [0, M); out row = b*(2*Ntok) + branch*Ntok + n
// ---------------------------------------------------------------------------
__global__ void layernorm_kernel(const float* __restrict__ O,
                                 const float* __restrict__ g,
                                 const float* __restrict__ beta,
                                 float* __restrict__ out,
                                 int L, int Ntok, int branch) {
    int m = blockIdx.x;
    int tid = threadIdx.x;
    const float* row = O + (size_t)m * L;
    float s = 0.f, s2 = 0.f;
    for (int i = tid; i < L; i += 256) {
        float v = row[i];
        s += v; s2 += v * v;
    }
    __shared__ float rs[256], rq[256];
    rs[tid] = s; rq[tid] = s2;
    __syncthreads();
    for (int off = 128; off > 0; off >>= 1) {
        if (tid < off) { rs[tid] += rs[tid + off]; rq[tid] += rq[tid + off]; }
        __syncthreads();
    }
    float mu  = rs[0] / (float)L;
    float var = rq[0] / (float)L - mu * mu;
    float inv = rsqrtf(var + 1e-5f);
    int b = m / Ntok;
    int n = m % Ntok;
    float* orow = out + ((size_t)b * (2 * Ntok) + (size_t)branch * Ntok + n) * L;
    for (int i = tid; i < L; i += 256) {
        float v = row[i];
        orow[i] = (v - mu) * inv * g[i] + beta[i];
    }
}

// ---------------------------------------------------------------------------
// Host-side orchestration
// ---------------------------------------------------------------------------
extern "C" void kernel_launch(void* const* d_in, const int* in_sizes, int n_in,
                              void* d_out, int out_size, void* d_ws, size_t ws_size,
                              hipStream_t stream) {
    (void)in_sizes; (void)n_in; (void)out_size; (void)ws_size;

    const float* img   = (const float*)d_in[0];
    const float* feat  = (const float*)d_in[1];
    const float* cb    = (const float*)d_in[2];
    const float* cw1   = (const float*)d_in[3];
    const float* cb1   = (const float*)d_in[4];
    const float* cw2   = (const float*)d_in[5];
    const float* cb2   = (const float*)d_in[6];
    const float* cw3   = (const float*)d_in[7];
    const float* cb3   = (const float*)d_in[8];
    const float* cw4   = (const float*)d_in[9];
    const float* cb4   = (const float*)d_in[10];
    const float* gw1   = (const float*)d_in[11];
    const float* gb1   = (const float*)d_in[12];
    const float* gw2   = (const float*)d_in[13];
    const float* gb2   = (const float*)d_in[14];
    const float* gg    = (const float*)d_in[15];
    const float* gbeta = (const float*)d_in[16];
    const float* uw1   = (const float*)d_in[17];
    const float* ub1   = (const float*)d_in[18];
    const float* uw2   = (const float*)d_in[19];
    const float* ub2   = (const float*)d_in[20];
    const float* ug    = (const float*)d_in[21];
    const float* ubeta = (const float*)d_in[22];
    float* out = (float*)d_out;

    const int Bn = 8, D = 256, Kcb = 8192, L = 2048, Ntok = 1024, HD = 8192, S = 1152;
    const int M = Bn * Ntok;  // 8192 tokens

    char* ws = (char*)d_ws;
    float*              x1    = (float*)(ws + 0);            // 8*64*128*128
    float*              x2    = (float*)(ws + 33554432);     // 8*128*64*64
    float*              x3    = (float*)(ws + 50331648);     // 8*256*32*32
    float*              x4    = (float*)(ws + 58720256);     // 8*256*32*32
    unsigned short*     embB  = (unsigned short*)(ws + 67108864);   // [8192,256]
    unsigned short*     cbB   = (unsigned short*)(ws + 71303168);   // [8192,256]
    float*              cbsq  = (float*)(ws + 75497472);            // [8192]
    unsigned long long* amin  = (unsigned long long*)(ws + 75530240); // [8192]
    unsigned short*     quantB= (unsigned short*)(ws + 75595776);   // [8192,256]
    unsigned short*     featB = (unsigned short*)(ws + 79790080);   // [8192,1152]
    unsigned short*     w1B   = (unsigned short*)(ws + 98664448);   // up to [8192,1152]
    unsigned short*     w2B   = (unsigned short*)(ws + 117538816);  // [2048,8192]
    unsigned short*     Hb    = (unsigned short*)(ws + 151093248);  // [8192,8192]
    float*              Obuf  = (float*)(ws + 285310976);           // [8192,2048]

    // ---- VQ conv encoder ----
    {
        int tot = Bn * 64 * 128 * 128;
        conv2d_kernel<<<(tot + 255) / 256, 256, 0, stream>>>(
            img, cw1, cb1, x1, Bn, 3, 256, 256, 64, 128, 128, 4, 4, 2, 1, 1);
    }
    {
        int tot = Bn * 128 * 64 * 64;
        conv2d_kernel<<<(tot + 255) / 256, 256, 0, stream>>>(
            x1, cw2, cb2, x2, Bn, 64, 128, 128, 128, 64, 64, 4, 4, 2, 1, 1);
    }
    {
        int tot = Bn * 256 * 32 * 32;
        conv2d_kernel<<<(tot + 255) / 256, 256, 0, stream>>>(
            x2, cw3, cb3, x3, Bn, 128, 64, 64, 256, 32, 32, 4, 4, 2, 1, 1);
    }
    {
        int tot = Bn * 256 * 32 * 32;
        conv2d_kernel<<<(tot + 255) / 256, 256, 0, stream>>>(
            x3, cw4, cb4, x4, Bn, 256, 32, 32, 256, 32, 32, 3, 3, 1, 1, 0);
    }

    // ---- quantize: cast, distances via WMMA, fused argmin, gather ----
    emb_transpose_cast_kernel<<<(M * D + 255) / 256, 256, 0, stream>>>(x4, embB, Bn, D, 1024);
    cast_bf16_kernel<<<(Kcb * D + 255) / 256, 256, 0, stream>>>(cb, cbB, Kcb * D);
    cbsq_kernel<<<(Kcb + 255) / 256, 256, 0, stream>>>(cb, cbsq, Kcb, D);
    amin_init_kernel<<<(M + 255) / 256, 256, 0, stream>>>(amin, M);
    {
        dim3 grid(Kcb / 128, M / 128);
        gemm_wmma_kernel<2><<<grid, 256, 0, stream>>>(
            embB, cbB, nullptr, nullptr, nullptr, cbsq, amin, M, Kcb, D);
    }
    gather_cast_kernel<<<(M * D + 255) / 256, 256, 0, stream>>>(amin, cb, quantB, M, D);

    // ---- generation branch adapter ----
    cast_bf16_kernel<<<(HD * D + 255) / 256, 256, 0, stream>>>(gw1, w1B, HD * D);
    {
        dim3 grid(HD / 128, M / 128);
        gemm_wmma_kernel<1><<<grid, 256, 0, stream>>>(
            quantB, w1B, gb1, nullptr, Hb, nullptr, nullptr, M, HD, D);
    }
    cast_bf16_kernel<<<(L * HD + 255) / 256, 256, 0, stream>>>(gw2, w2B, L * HD);
    {
        dim3 grid(L / 128, M / 128);
        gemm_wmma_kernel<0><<<grid, 256, 0, stream>>>(
            Hb, w2B, gb2, Obuf, nullptr, nullptr, nullptr, M, L, HD);
    }
    layernorm_kernel<<<M, 256, 0, stream>>>(Obuf, gg, gbeta, out, L, Ntok, 0);

    // ---- understanding branch adapter ----
    cast_bf16_kernel<<<(M * S + 255) / 256, 256, 0, stream>>>(feat, featB, M * S);
    cast_bf16_kernel<<<(HD * S + 255) / 256, 256, 0, stream>>>(uw1, w1B, HD * S);
    {
        dim3 grid(HD / 128, M / 128);
        gemm_wmma_kernel<1><<<grid, 256, 0, stream>>>(
            featB, w1B, ub1, nullptr, Hb, nullptr, nullptr, M, HD, S);
    }
    cast_bf16_kernel<<<(L * HD + 255) / 256, 256, 0, stream>>>(uw2, w2B, L * HD);
    {
        dim3 grid(L / 128, M / 128);
        gemm_wmma_kernel<0><<<grid, 256, 0, stream>>>(
            Hb, w2B, ub2, Obuf, nullptr, nullptr, nullptr, M, L, HD);
    }
    layernorm_kernel<<<M, 256, 0, stream>>>(Obuf, ug, ubeta, out, L, Ntok, 1);
}
